// SigLipAttention_8735963480392
// MI455X (gfx1250) — matrix operand order repaired and checked by reading, not compiled, not run on hardware
//
#include <hip/hip_runtime.h>
#include <hip/hip_bf16.h>
#include <math.h>

typedef __attribute__((ext_vector_type(16))) _Float16 v16h;
typedef __attribute__((ext_vector_type(8)))  float    v8f;

static __device__ inline v8f wmma_f16(v16h a, v16h b, v8f c) {
    return __builtin_amdgcn_wmma_f32_16x16x32_f16(
        false, a, false, b, (short)0, c, false, false);
}

// ---------------------------------------------------------------------------
// Kernel 1: fused QKV projection. out = x @ W + bias, cast to f16.
//   z==0 -> Q  stored [bh][s][hd]
//   z==1 -> Kt stored [bh][hd][s]   (pre-transposed for scores B-fragments)
//   z==2 -> V  stored [bh][s][hd]
// Tile: 128(M) x 128(N), K-step 32, 8 waves (2x4), each wave 4x2 WMMA tiles.
// ---------------------------------------------------------------------------
__global__ __launch_bounds__(256) void qkv_proj_kernel(
    const float* __restrict__ x,
    const float* __restrict__ Wq, const float* __restrict__ bq,
    const float* __restrict__ Wk, const float* __restrict__ bk,
    const float* __restrict__ Wv, const float* __restrict__ bv,
    _Float16* __restrict__ qbuf, _Float16* __restrict__ ktbuf,
    _Float16* __restrict__ vbuf)
{
    constexpr int D = 1024;
    __shared__ _Float16 xT[128][32];   // A-staging: [m][k]
    __shared__ _Float16 wT[32][128];   // B-staging: [k][n]

    const int z = blockIdx.z;
    const float* W    = (z == 0) ? Wq : (z == 1) ? Wk : Wv;
    const float* bias = (z == 0) ? bq : (z == 1) ? bk : bv;

    const int tid   = threadIdx.x;
    const int lane  = tid & 31;
    const int wv    = tid >> 5;        // wave 0..7
    const int mWave = (wv >> 2) * 64;  // 2 waves along M
    const int nWave = (wv & 3) * 32;   // 4 waves along N
    const int rowBlk = blockIdx.x * 128;
    const int colBlk = blockIdx.y * 128;

    const int l15   = lane & 15;
    const int lhi   = lane >> 4;
    const int khalf = lhi * 8;

    v8f acc[4][2];
    for (int tm = 0; tm < 4; ++tm)
        for (int tn = 0; tn < 2; ++tn)
            acc[tm][tn] = (v8f){0.f, 0.f, 0.f, 0.f, 0.f, 0.f, 0.f, 0.f};

    for (int kk = 0; kk < D; kk += 32) {
        // stage x tile 128x32 (f32 -> f16)
        for (int e = 0; e < 16; ++e) {
            int idx = tid * 16 + e;          // 0..4095
            int r = idx >> 5, c = idx & 31;
            xT[r][c] = (_Float16)x[(size_t)(rowBlk + r) * D + kk + c];
        }
        // stage W tile 32x128
        for (int e = 0; e < 16; ++e) {
            int idx = tid * 16 + e;          // 0..4095
            int r = idx >> 7, c = idx & 127;
            wT[r][c] = (_Float16)W[(size_t)(kk + r) * D + colBlk + c];
        }
        __syncthreads();

        // A fragments: 16x32 f16, lane-half splits K into 8-blocks
        v16h aF[4];
        for (int tm = 0; tm < 4; ++tm) {
            const int row = mWave + tm * 16 + l15;
            v16h a;
            for (int v = 0; v < 8; ++v) {
                int k0 = (v < 4) ? (khalf + 2 * v) : (16 + khalf + 2 * (v - 4));
                a[2 * v]     = xT[row][k0];
                a[2 * v + 1] = xT[row][k0 + 1];
            }
            aF[tm] = a;
        }
        // B fragments: 32x16, lane = N-col, K in same order as A
        v16h bF[2];
        for (int tn = 0; tn < 2; ++tn) {
            const int c0 = nWave + tn * 16;
            v16h b;
            for (int v = 0; v < 8; ++v) {
                int k0 = (v < 4) ? (khalf + 2 * v) : (16 + khalf + 2 * (v - 4));
                b[2 * v]     = wT[k0][c0 + l15];
                b[2 * v + 1] = wT[k0 + 1][c0 + l15];
            }
            bF[tn] = b;
        }
        for (int tm = 0; tm < 4; ++tm)
            for (int tn = 0; tn < 2; ++tn)
                acc[tm][tn] = wmma_f16(aF[tm], bF[tn], acc[tm][tn]);
        __syncthreads();
    }

    // epilogue: bias add, f16 cast, scatter into attention layouts.
    // Layout branch hoisted out of the store loops.
    if (z == 1) {
        for (int tn = 0; tn < 2; ++tn) {
            const int n  = colBlk + nWave + tn * 16 + l15;
            const float bn = bias[n];
            const int hh = n >> 6, d = n & 63;
            for (int tm = 0; tm < 4; ++tm)
                for (int r = 0; r < 8; ++r) {
                    int m = rowBlk + mWave + tm * 16 + r + 8 * lhi;
                    int b = m >> 10, s = m & 1023;
                    ktbuf[((size_t)(b * 16 + hh) * 64 + d) * 1024 + s] =
                        (_Float16)(acc[tm][tn][r] + bn);
                }
        }
    } else {
        _Float16* out = (z == 0) ? qbuf : vbuf;
        for (int tn = 0; tn < 2; ++tn) {
            const int n  = colBlk + nWave + tn * 16 + l15;
            const float bn = bias[n];
            const int hh = n >> 6, d = n & 63;
            for (int tm = 0; tm < 4; ++tm)
                for (int r = 0; r < 8; ++r) {
                    int m = rowBlk + mWave + tm * 16 + r + 8 * lhi;
                    int b = m >> 10, s = m & 1023;
                    out[((size_t)(b * 16 + hh) * 1024 + s) * 64 + d] =
                        (_Float16)(acc[tm][tn][r] + bn);
                }
        }
    }
}

// ---------------------------------------------------------------------------
// Kernel 2: per (b,h) attention. Kt [64][1024] + V [1024][64] in LDS (f16),
// filled with GLOBAL_LOAD_ASYNC_TO_LDS_B128 (ASYNCcnt path, no VGPR bounce).
// 8 waves x 8 strips of 16 query rows. Pass 1: online softmax stats.
// Pass 2: recompute scores, NT-store normalized weights to global, P via
// per-wave LDS staging -> A fragment, O += P x V with WMMA.
// NOTE: no static __shared__ here, so dynamic LDS starts at byte offset 0.
// ---------------------------------------------------------------------------
__global__ __launch_bounds__(256) void attn_kernel(
    const _Float16* __restrict__ qbuf,
    const _Float16* __restrict__ ktbuf,
    const _Float16* __restrict__ vbuf,
    float* __restrict__ attn_out,      // [B, S, D]
    float* __restrict__ weights_out)   // [B, H, S, S]
{
    extern __shared__ _Float16 smem[];
    _Float16* Kt = smem;                    // 64 x 1024    (LDS bytes 0..131071)
    _Float16* Vt = smem + 64 * 1024;        // 1024 x 64    (LDS bytes 131072..)
    _Float16* pb = smem + 2 * 64 * 1024;    // 8 waves x 16 x 32 P staging

    const int bh = blockIdx.x;              // b*16 + h
    const int b  = bh >> 4, h = bh & 15;
    const int tid  = threadIdx.x;
    const int lane = tid & 31;
    const int wv   = tid >> 5;
    const int l15  = lane & 15;
    const int lhi  = lane >> 4;
    const int khalf = lhi * 8;
    const float scale = 0.125f;             // 1/sqrt(64)

    // ---- LDS fill: 128 KB Kt + 128 KB V ----
#if defined(__gfx1250__)
    {
        const unsigned long long gk =
            (unsigned long long)(ktbuf + (size_t)bh * 65536);
        const unsigned long long gv =
            (unsigned long long)(vbuf + (size_t)bh * 65536);
        const unsigned ldsV = 131072u;      // dynamic LDS base == 0, Vt offset
        for (int i = tid; i < 8192; i += 256) {
            const unsigned boff = (unsigned)i * 16u;   // byte offset, 16B/lane
            asm volatile("global_load_async_to_lds_b128 %0, %1, %2 offset:0"
                         :: "v"(boff), "v"(boff), "s"(gk) : "memory");
            asm volatile("global_load_async_to_lds_b128 %0, %1, %2 offset:0"
                         :: "v"(ldsV + boff), "v"(boff), "s"(gv) : "memory");
        }
        asm volatile("s_wait_asynccnt 0x0" ::: "memory");
    }
#else
    {
        const uint4* gk = (const uint4*)(ktbuf + (size_t)bh * 65536);
        const uint4* gv = (const uint4*)(vbuf  + (size_t)bh * 65536);
        uint4* lk = (uint4*)Kt;
        uint4* lv = (uint4*)Vt;
        for (int i = tid; i < 8192; i += 256) { lk[i] = gk[i]; lv[i] = gv[i]; }
    }
#endif
    __syncthreads();

    for (int si = 0; si < 8; ++si) {
        const int rowBase = (si * 8 + wv) * 16;

        // Q A-fragments for this 16-row strip (hd=64 -> 2 k-steps of 32)
        v16h aq[2];
        const _Float16* qrow = qbuf + ((size_t)bh * 1024 + rowBase + l15) * 64;
        for (int ks = 0; ks < 2; ++ks) {
            v16h a;
            for (int v = 0; v < 8; ++v) {
                int k0 = ks * 32 +
                         ((v < 4) ? (khalf + 2 * v) : (16 + khalf + 2 * (v - 4)));
                a[2 * v]     = qrow[k0];
                a[2 * v + 1] = qrow[k0 + 1];
            }
            aq[ks] = a;
        }

        // ---- pass 1: online softmax stats (per-lane, then width-16 merge)
        float mrow[8], srow[8];
        for (int r = 0; r < 8; ++r) { mrow[r] = -3.0e38f; srow[r] = 0.f; }

        for (int nt = 0; nt < 64; ++nt) {
            v8f c = (v8f){0.f, 0.f, 0.f, 0.f, 0.f, 0.f, 0.f, 0.f};
            for (int ks = 0; ks < 2; ++ks) {
                const _Float16* kr = Kt + ks * 32 * 1024 + nt * 16 + l15;
                v16h bf;
                for (int v = 0; v < 8; ++v) {
                    int k0 = (v < 4) ? (khalf + 2 * v) : (16 + khalf + 2 * (v - 4));
                    bf[2 * v]     = kr[k0 * 1024];
                    bf[2 * v + 1] = kr[(k0 + 1) * 1024];
                }
                c = wmma_f16(aq[ks], bf, c);
            }
            for (int r = 0; r < 8; ++r) {
                float xv = c[r] * scale;
                float nm = fmaxf(mrow[r], xv);
                srow[r] = srow[r] * __expf(mrow[r] - nm) + __expf(xv - nm);
                mrow[r] = nm;
            }
        }
        for (int off = 1; off < 16; off <<= 1)
            for (int r = 0; r < 8; ++r) {
                float om = __shfl_xor(mrow[r], off, 16);
                float os = __shfl_xor(srow[r], off, 16);
                float nm = fmaxf(mrow[r], om);
                srow[r] = srow[r] * __expf(mrow[r] - nm) + os * __expf(om - nm);
                mrow[r] = nm;
            }
        float inv_s[8];
        for (int r = 0; r < 8; ++r) inv_s[r] = 1.0f / srow[r];

        // ---- pass 2: weights out (non-temporal) + O = P x V
        v8f oc[4];
        for (int dt = 0; dt < 4; ++dt)
            oc[dt] = (v8f){0.f, 0.f, 0.f, 0.f, 0.f, 0.f, 0.f, 0.f};
        _Float16* pw = pb + wv * 512;   // per-wave [16][32]

        for (int nt = 0; nt < 64; ++nt) {
            v8f c = (v8f){0.f, 0.f, 0.f, 0.f, 0.f, 0.f, 0.f, 0.f};
            for (int ks = 0; ks < 2; ++ks) {
                const _Float16* kr = Kt + ks * 32 * 1024 + nt * 16 + l15;
                v16h bf;
                for (int v = 0; v < 8; ++v) {
                    int k0 = (v < 4) ? (khalf + 2 * v) : (16 + khalf + 2 * (v - 4));
                    bf[2 * v]     = kr[k0 * 1024];
                    bf[2 * v + 1] = kr[(k0 + 1) * 1024];
                }
                c = wmma_f16(aq[ks], bf, c);
            }
            for (int r = 0; r < 8; ++r) {
                float wgt = __expf(c[r] * scale - mrow[r]) * inv_s[r];
                // write-once 512MB stream: non-temporal store
                __builtin_nontemporal_store(
                    wgt,
                    &weights_out[((size_t)bh * 1024 + rowBase + r + 8 * lhi) *
                                     1024 + nt * 16 + l15]);
                pw[(r + 8 * lhi) * 32 + (nt & 1) * 16 + l15] = (_Float16)wgt;
            }
            if (nt & 1) {
                // P A-fragment from per-wave LDS (intra-wave DS is ordered)
                v16h ap;
                for (int v = 0; v < 8; ++v) {
                    int k0 = (v < 4) ? (khalf + 2 * v)
                                     : (16 + khalf + 2 * (v - 4));
                    ap[2 * v]     = pw[l15 * 32 + k0];
                    ap[2 * v + 1] = pw[l15 * 32 + k0 + 1];
                }
                const _Float16* vr = Vt + (nt >> 1) * 32 * 64 + l15;
                for (int dt = 0; dt < 4; ++dt) {
                    v16h bf;
                    for (int v = 0; v < 8; ++v) {
                        int k0 = (v < 4) ? (khalf + 2 * v)
                                         : (16 + khalf + 2 * (v - 4));
                        bf[2 * v]     = vr[k0 * 64 + dt * 16];
                        bf[2 * v + 1] = vr[(k0 + 1) * 64 + dt * 16];
                    }
                    oc[dt] = wmma_f16(ap, bf, oc[dt]);
                }
            }
        }
        // store O: attn[b, s, h*64 + d] (write-once, non-temporal)
        for (int dt = 0; dt < 4; ++dt)
            for (int r = 0; r < 8; ++r)
                __builtin_nontemporal_store(
                    oc[dt][r],
                    &attn_out[((size_t)b * 1024 + rowBase + r + 8 * lhi) * 1024 +
                              h * 64 + dt * 16 + l15]);
    }
}

// ---------------------------------------------------------------------------
extern "C" void kernel_launch(void* const* d_in, const int* in_sizes, int n_in,
                              void* d_out, int out_size, void* d_ws,
                              size_t ws_size, hipStream_t stream) {
    const float* x  = (const float*)d_in[0];
    const float* Wq = (const float*)d_in[1];
    const float* bq = (const float*)d_in[2];
    const float* Wk = (const float*)d_in[3];
    const float* bk = (const float*)d_in[4];
    const float* Wv = (const float*)d_in[5];
    const float* bv = (const float*)d_in[6];

    // workspace: Q, Kt, V in f16, 8M halves each (48 MB total)
    _Float16* qbuf  = (_Float16*)d_ws;
    _Float16* ktbuf = qbuf + (size_t)8 * 1024 * 1024;
    _Float16* vbuf  = ktbuf + (size_t)8 * 1024 * 1024;

    float* attn_out    = (float*)d_out;                          // 8*1024*1024
    float* weights_out = attn_out + (size_t)8 * 1024 * 1024;     // 8*16*1024*1024

    dim3 g1(64, 8, 3);   // M/128, N/128, {Q,K,V}
    qkv_proj_kernel<<<g1, 256, 0, stream>>>(x, Wq, bq, Wk, bk, Wv, bv,
                                            qbuf, ktbuf, vbuf);

    const size_t lds_bytes = 2u * 64 * 1024 * sizeof(_Float16)   // Kt + V
                           + 8u * 512 * sizeof(_Float16);        // P staging
    (void)hipFuncSetAttribute((const void*)attn_kernel,
                              hipFuncAttributeMaxDynamicSharedMemorySize,
                              (int)lds_bytes);
    attn_kernel<<<128, 256, lds_bytes, stream>>>(qbuf, ktbuf, vbuf,
                                                 attn_out, weights_out);
}
